// myTFRNNModel_71708773974319
// MI455X (gfx1250) — matrix operand order repaired and checked by reading, not compiled, not run on hardware
//
#include <hip/hip_runtime.h>

// CDNA5 / gfx1250 WMMA fragment types
typedef __attribute__((ext_vector_type(16))) _Float16 v16h;
typedef __attribute__((ext_vector_type(8)))  float    v8f;

#define BB    256
#define SS    2048
#define VOCAB 10
#define EE    32
#define HH    64
#define NOUT  10
#define ROWS  16   // batch rows per workgroup == WMMA M

#if __has_builtin(__builtin_amdgcn_tanhf)
#define FAST_TANH(x) __builtin_amdgcn_tanhf(x)
#else
#define FAST_TANH(x) tanhf(x)
#endif

// One wave32 per block; each block owns 16 batch rows for the whole sequence.
// Fragment layout conventions used (per CDNA5 ISA §7.12.2):
//   A (16x32 f16): lane l holds row M=l&15; half=l>>4;
//                  elem e: K = half*16 + e
//   B (32x16 f16): lane l holds col N=l&15; elem e: K = half*16 + e
//   C/D (16x16 f32): lane l, vgpr r -> (M = (l>>4)*8 + r, N = l&15)
__launch_bounds__(32)
__global__ void rnn_fused_kernel(const int*   __restrict__ num1,
                                 const int*   __restrict__ num2,
                                 const float* __restrict__ embed,
                                 const float* __restrict__ Wx,
                                 const float* __restrict__ Wh,
                                 const float* __restrict__ bias,
                                 const float* __restrict__ Wd,
                                 const float* __restrict__ bd,
                                 float*       __restrict__ out) {
  __shared__ float T1[VOCAB * HH];      // embed @ Wx[:32] + b
  __shared__ float T2[VOCAB * HH];      // embed @ Wx[32:]
  __shared__ float hbuf[ROWS * HH];     // h_t bounce buffer (D-fmt -> A-fmt)
  __shared__ int   idxbuf[2][ROWS];

  const int lane  = threadIdx.x;        // 0..31
  const int half  = lane >> 4;          // 0/1
  const int lan16 = lane & 15;
  const int row0  = blockIdx.x * ROWS;  // first batch row of this tile

  // ---- one-time: fold embedding + input projection into 10x64 LUTs ----
  for (int i = lane; i < 2 * VOCAB * HH; i += 32) {
    const int tab  = i / (VOCAB * HH);
    const int rem  = i % (VOCAB * HH);
    const int v    = rem / HH;
    const int hcol = rem % HH;
    float s = (tab == 0) ? bias[hcol] : 0.0f;
#pragma unroll 8
    for (int e = 0; e < EE; ++e)
      s += embed[v * EE + e] * Wx[(tab * EE + e) * HH + hcol];
    if (tab == 0) T1[v * HH + hcol] = s;
    else          T2[v * HH + hcol] = s;
  }

  // ---- one-time: Wh / Wd as persistent B-format f16 fragments ----
  v16h bWh[2][4];                       // [k-chunk][n-tile]
#pragma unroll
  for (int kc = 0; kc < 2; ++kc)
#pragma unroll
    for (int nt = 0; nt < 4; ++nt) {
      const int N = nt * 16 + lan16;
#pragma unroll
      for (int e = 0; e < 16; ++e) {
        const int K = kc * 32 + half * 16 + e;
        bWh[kc][nt][e] = (_Float16)Wh[K * HH + N];
      }
    }
  v16h bWd[2];                          // cols 10..15 zero-padded
#pragma unroll
  for (int kc = 0; kc < 2; ++kc)
#pragma unroll
    for (int e = 0; e < 16; ++e) {
      const int K = kc * 32 + half * 16 + e;
      bWd[kc][e] = (lan16 < NOUT) ? (_Float16)Wd[K * NOUT + lan16] : (_Float16)0.0f;
    }
  const float bdv = (lan16 < NOUT) ? bd[lan16] : 0.0f;

  v16h aH[2] = {};                      // h_{t-1} A-fragments; h0 = 0

  __syncthreads();                      // LUTs ready (single-wave WG: ~free)

  for (int t = 0; t < SS; ++t) {
    // indices for this step: lanes 0-15 fetch num1, lanes 16-31 fetch num2
    if (half == 0) idxbuf[0][lan16] = num1[(row0 + lan16) * SS + t];
    else           idxbuf[1][lan16] = num2[(row0 + lan16) * SS + t];
    if (t + 16 < SS) {                  // gfx1250 global_prefetch_b8
      const int* p = (half == 0) ? &num1[(row0 + lan16) * SS + t + 16]
                                 : &num2[(row0 + lan16) * SS + t + 16];
      __builtin_prefetch(p, 0, 0);
    }
    __syncthreads();

    // acc = xw_t in C/D format (f32), gathered from the LUTs
    v8f acc[4];
#pragma unroll
    for (int nt = 0; nt < 4; ++nt) {
      const int N = nt * 16 + lan16;
#pragma unroll
      for (int r = 0; r < 8; ++r) {
        const int M = half * 8 + r;
        acc[nt][r] = T1[idxbuf[0][M] * HH + N] + T2[idxbuf[1][M] * HH + N];
      }
    }

    // h_t_pre = xw_t + h_{t-1} @ Wh   (8 chained WMMAs, f32 accumulate)
#pragma unroll
    for (int nt = 0; nt < 4; ++nt) {
      acc[nt] = __builtin_amdgcn_wmma_f32_16x16x32_f16(
          false, aH[0], false, bWh[0][nt], (short)0, acc[nt], false, false);
      acc[nt] = __builtin_amdgcn_wmma_f32_16x16x32_f16(
          false, aH[1], false, bWh[1][nt], (short)0, acc[nt], false, false);
    }

    __syncthreads();                    // WAR: prior A-rebuild reads of hbuf
    // h_t = tanh(.)  -> LDS in row-major [16][64]
#pragma unroll
    for (int nt = 0; nt < 4; ++nt) {
      const int N = nt * 16 + lan16;
#pragma unroll
      for (int r = 0; r < 8; ++r)
        hbuf[(half * 8 + r) * HH + N] = FAST_TANH(acc[nt][r]);
    }
    __syncthreads();

    // rebuild A-format h_t fragments (feed next step AND the output head)
#pragma unroll
    for (int kc = 0; kc < 2; ++kc)
#pragma unroll
      for (int e = 0; e < 16; ++e) {
        const int K = kc * 32 + half * 16 + e;
        aH[kc][e] = (_Float16)hbuf[lan16 * HH + K];
      }

    // logits_t = h_t @ Wd + bd   (2 WMMAs, reuse aH fragments)
    v8f dacc;
#pragma unroll
    for (int r = 0; r < 8; ++r) dacc[r] = bdv;
    dacc = __builtin_amdgcn_wmma_f32_16x16x32_f16(
        false, aH[0], false, bWd[0], (short)0, dacc, false, false);
    dacc = __builtin_amdgcn_wmma_f32_16x16x32_f16(
        false, aH[1], false, bWd[1], (short)0, dacc, false, false);

    if (lan16 < NOUT) {
#pragma unroll
      for (int r = 0; r < 8; ++r) {
        const int M = half * 8 + r;
        out[((size_t)(row0 + M) * SS + t) * NOUT + lan16] = dacc[r];
      }
    }
  }
}

extern "C" void kernel_launch(void* const* d_in, const int* in_sizes, int n_in,
                              void* d_out, int out_size, void* d_ws, size_t ws_size,
                              hipStream_t stream) {
  (void)in_sizes; (void)n_in; (void)d_ws; (void)ws_size; (void)out_size;
  const int*   num1  = (const int*)  d_in[0];
  const int*   num2  = (const int*)  d_in[1];
  const float* embed = (const float*)d_in[2];
  const float* Wx    = (const float*)d_in[3];
  const float* Wh    = (const float*)d_in[4];
  const float* b     = (const float*)d_in[5];
  const float* Wd    = (const float*)d_in[6];
  const float* bd    = (const float*)d_in[7];
  float* out = (float*)d_out;

  dim3 grid(BB / ROWS);   // 16 workgroups, one 16-row tile each
  dim3 block(32);         // one wave32 per workgroup: zero-cost barriers
  rnn_fused_kernel<<<grid, block, 0, stream>>>(num1, num2, embed, Wx, Wh, b,
                                               Wd, bd, out);
}
